// LanguageModel_15891378995633
// MI455X (gfx1250) — matrix operand order repaired and checked by reading, not compiled, not run on hardware
//
#include <hip/hip_runtime.h>
#include <hip/hip_bf16.h>

typedef __attribute__((ext_vector_type(16))) _Float16 v16h;
typedef __attribute__((ext_vector_type(8)))  float    v8f;
typedef unsigned int u32x4 __attribute__((ext_vector_type(4)));
typedef int          i32x4 __attribute__((ext_vector_type(4)));
typedef int          i32x8 __attribute__((ext_vector_type(8)));

constexpr int Bn = 8, Sn = 256, Hn = 768, Vn = 21128, Ln = 12, NHn = 12, DHn = 64, FFn = 3072;
constexpr int SEPtok = 102, PADtok = 0;
constexpr int Mtok = Bn * Sn;   // 2048 token rows

#if defined(__has_builtin)
#if __has_builtin(__builtin_amdgcn_tensor_load_to_lds)
#define HAVE_TDM 1
#endif
#endif

#ifdef HAVE_TDM
// Issue a TDM load of a BMxBK f16 tile (row stride K elements) into LDS.
__device__ __forceinline__ void tdm_load_a(const _Float16* gsrc, unsigned lds,
                                           int M, int K, int BM, int BK)
{
    unsigned long long ga = (unsigned long long)(const void*)gsrc;
    u32x4 g0;
    g0[0] = 1u;                                   // count=1 (valid), user mode
    g0[1] = lds;                                  // lds_addr
    g0[2] = (unsigned)(ga & 0xFFFFFFFFu);         // global_addr[31:0]
    g0[3] = ((unsigned)(ga >> 32) & 0x01FFFFFFu) | (2u << 30);  // addr[56:32] | type=2
    i32x8 g1;
    g1[0] = (int)(1u << 16);                      // wg_mask=0, data_size=1 (2B)
    g1[1] = (int)(((unsigned)K & 0xFFFFu) << 16); // tensor_dim0[15:0]
    g1[2] = (int)((((unsigned)K >> 16) & 0xFFFFu) | (((unsigned)M & 0xFFFFu) << 16));
    g1[3] = (int)((((unsigned)M >> 16) & 0xFFFFu) | ((unsigned)BK << 16)); // tile_dim0
    g1[4] = (int)BM;                              // tile_dim1, tile_dim2=0
    g1[5] = (int)K;                               // tensor_dim0_stride[31:0]
    g1[6] = 0; g1[7] = 0;
    i32x4 g2 = {0, 0, 0, 0}, g3 = {0, 0, 0, 0};   // 2D tensor: groups 2/3 unused
#if defined(__clang_major__) && (__clang_major__ >= 23)
    i32x8 g4 = {0, 0, 0, 0, 0, 0, 0, 0};
    __builtin_amdgcn_tensor_load_to_lds(g0, g1, g2, g3, g4, 0);
#else
    __builtin_amdgcn_tensor_load_to_lds(g0, g1, g2, g3, 0);
#endif
}
#endif

// ---------------------------------------------------------------------------
// sep_pos[b] = index of first SEP + 1, else S
// ---------------------------------------------------------------------------
__global__ __launch_bounds__(32) void k_sep(const int* __restrict__ x, int* __restrict__ sep)
{
    int b = threadIdx.x;
    if (b < Bn) {
        int sp = Sn;
        for (int i = 0; i < Sn; ++i) {
            if (x[b * Sn + i] == SEPtok) { sp = i + 1; break; }
        }
        sep[b] = sp;
    }
}

// ---------------------------------------------------------------------------
// Embedding gather + LayerNorm -> h (f32) and h16 (f16)
// ---------------------------------------------------------------------------
__global__ __launch_bounds__(256) void k_embed(const int* __restrict__ x,
                                               const float* __restrict__ we,
                                               const float* __restrict__ pe,
                                               const float* __restrict__ te,
                                               const float* __restrict__ lns,
                                               const float* __restrict__ lnb,
                                               float* __restrict__ hout,
                                               _Float16* __restrict__ hout16)
{
    __shared__ float red[256];
    int row = blockIdx.x;
    int ss  = row % Sn;
    int tok = x[row];
    int t = threadIdx.x;
    float vals[3];
    float sum = 0.f;
#pragma unroll
    for (int i = 0; i < 3; ++i) {
        int cc = t + i * 256;
        float v = we[(size_t)tok * Hn + cc] + pe[(size_t)ss * Hn + cc] + te[cc];
        vals[i] = v; sum += v;
    }
    red[t] = sum; __syncthreads();
    for (int st = 128; st > 0; st >>= 1) { if (t < st) red[t] += red[t + st]; __syncthreads(); }
    float mean = red[0] / Hn; __syncthreads();
    float var = 0.f;
#pragma unroll
    for (int i = 0; i < 3; ++i) { float d = vals[i] - mean; var += d * d; }
    red[t] = var; __syncthreads();
    for (int st = 128; st > 0; st >>= 1) { if (t < st) red[t] += red[t + st]; __syncthreads(); }
    float rstd = rsqrtf(red[0] / Hn + 1e-12f);
#pragma unroll
    for (int i = 0; i < 3; ++i) {
        int cc = t + i * 256;
        float o = (vals[i] - mean) * rstd * lns[cc] + lnb[cc];
        hout[(size_t)row * Hn + cc] = o;
        hout16[(size_t)row * Hn + cc] = (_Float16)o;
    }
}

// ---------------------------------------------------------------------------
// h = LN(h + delta) -> h (f32), h16 (f16)
// ---------------------------------------------------------------------------
__global__ __launch_bounds__(256) void k_add_ln(const float* __restrict__ hin,
                                                const float* __restrict__ delta,
                                                const float* __restrict__ lns,
                                                const float* __restrict__ lnb,
                                                float* __restrict__ hout,
                                                _Float16* __restrict__ hout16)
{
    __shared__ float red[256];
    size_t row = blockIdx.x;
    int t = threadIdx.x;
    float vals[3];
    float sum = 0.f;
#pragma unroll
    for (int i = 0; i < 3; ++i) {
        int cc = t + i * 256;
        float v = hin[row * Hn + cc] + delta[row * Hn + cc];
        vals[i] = v; sum += v;
    }
    red[t] = sum; __syncthreads();
    for (int st = 128; st > 0; st >>= 1) { if (t < st) red[t] += red[t + st]; __syncthreads(); }
    float mean = red[0] / Hn; __syncthreads();
    float var = 0.f;
#pragma unroll
    for (int i = 0; i < 3; ++i) { float d = vals[i] - mean; var += d * d; }
    red[t] = var; __syncthreads();
    for (int st = 128; st > 0; st >>= 1) { if (t < st) red[t] += red[t + st]; __syncthreads(); }
    float rstd = rsqrtf(red[0] / Hn + 1e-12f);
#pragma unroll
    for (int i = 0; i < 3; ++i) {
        int cc = t + i * 256;
        float o = (vals[i] - mean) * rstd * lns[cc] + lnb[cc];
        hout[row * Hn + cc] = o;
        hout16[row * Hn + cc] = (_Float16)o;
    }
}

// ---------------------------------------------------------------------------
// GEMM: C[M,N] = A_f16[M,K] @ W_f32[K,N] + bias
// Block: 256 threads (8 wave32), tile 128x64; each wave owns a 32x32 output
// (2x2 WMMA register block -> 4 v_wmma per K-step, A/B fragments reused 2x).
// A tiles are streamed by the Tensor Data Mover into double-buffered LDS:
// DMA for tile k+1 is issued before computing tile k, and we only wait for
// TENSORcnt<=1 (in-order completion) so one DMA stays in flight across the
// whole W-staging + WMMA phase. W tiles are staged with fp32->f16 conversion.
// ---------------------------------------------------------------------------
__global__ __launch_bounds__(256) void k_gemm(const _Float16* __restrict__ A,
                                              const float* __restrict__ W,
                                              const float* __restrict__ bias,
                                              float* __restrict__ outF,
                                              _Float16* __restrict__ outH,
                                              int M, int N, int K)
{
    constexpr int BM = 128, BN = 64, BK = 32;
    __shared__ _Float16 As[2][BM][BK];    // 16 KB ping-pong (TDM target)
    __shared__ _Float16 Bt[BN][BK + 8];   // 5 KB, transposed [n][k], padded
    int tid  = threadIdx.x;
    int m0   = blockIdx.y * BM;
    int n0   = blockIdx.x * BN;
    int wid  = tid >> 5, lane = tid & 31;
    int wm   = wid & 3, wn = wid >> 2;       // 4x2 wave grid of 32x32 sub-tiles
    int grp  = lane >> 4, lr = lane & 15;
    v8f c00 = {}, c01 = {}, c10 = {}, c11 = {};

#ifdef HAVE_TDM
    if (tid < 32) {   // prologue: DMA first A tile into buffer 0
        tdm_load_a(A + (size_t)m0 * K, (unsigned)(unsigned long long)(void*)&As[0][0][0],
                   M, K, BM, BK);
    }
#endif
    for (int k0 = 0, step = 0; k0 < K; k0 += BK, ++step) {
        int buf = step & 1;
        __syncthreads();   // previous compute done before LDS buffers are reused
#ifdef HAVE_TDM
        if (tid < 32) {
            if (k0 + BK < K) {
                // issue DMA for next tile into the other buffer; keep it in flight
                tdm_load_a(A + (size_t)m0 * K + (k0 + BK),
                           (unsigned)(unsigned long long)(void*)&As[buf ^ 1][0][0],
                           M, K, BM, BK);
                __builtin_amdgcn_s_wait_tensorcnt(1);  // current tile landed
            } else {
                __builtin_amdgcn_s_wait_tensorcnt(0);  // drain on last step
            }
        }
#else
        {   // fallback: vectorized A tile load, 16 halves per thread
            int r  = tid >> 1;            // 0..127
            int cq = (tid & 1) * 16;      // 0 or 16
            const _Float16* src = A + (size_t)(m0 + r) * K + k0 + cq;
            *(float4*)(&As[buf][r][cq + 0]) = *(const float4*)(src + 0);
            *(float4*)(&As[buf][r][cq + 8]) = *(const float4*)(src + 8);
        }
#endif
        {   // W tile: 32 k-rows x 64 cols fp32 -> f16, transposed into Bt
            int kr = tid >> 3;            // 0..31
            int nc = (tid & 7) * 8;       // 0..56
            const float* wp = W + (size_t)(k0 + kr) * N + n0 + nc;
            float4 v0 = *(const float4*)(wp + 0);
            float4 v1 = *(const float4*)(wp + 4);
            Bt[nc + 0][kr] = (_Float16)v0.x;
            Bt[nc + 1][kr] = (_Float16)v0.y;
            Bt[nc + 2][kr] = (_Float16)v0.z;
            Bt[nc + 3][kr] = (_Float16)v0.w;
            Bt[nc + 4][kr] = (_Float16)v1.x;
            Bt[nc + 5][kr] = (_Float16)v1.y;
            Bt[nc + 6][kr] = (_Float16)v1.z;
            Bt[nc + 7][kr] = (_Float16)v1.w;
            if (k0 + BK < K)
                __builtin_prefetch(wp + (size_t)BK * N, 0, 0);
        }
        __syncthreads();

        v16h a0, a1, b0, b1;
        int r0 = wm * 32 + lr, r1 = r0 + 16;
        int q0 = wn * 32 + lr, q1 = q0 + 16;
#pragma unroll
        for (int j = 0; j < 16; ++j) {
            int vg = j >> 1;
            int ka = ((vg < 4) ? 0 : 16) + ((vg & 3) * 2) + (j & 1) + grp * 8;
            int kb = ((j < 8) ? j : j + 8) + grp * 8;
            a0[j] = As[buf][r0][ka];
            a1[j] = As[buf][r1][ka];
            b0[j] = Bt[q0][kb];
            b1[j] = Bt[q1][kb];
        }
        c00 = __builtin_amdgcn_wmma_f32_16x16x32_f16(false, a0, false, b0, (short)0, c00, false, false);
        c01 = __builtin_amdgcn_wmma_f32_16x16x32_f16(false, a0, false, b1, (short)0, c01, false, false);
        c10 = __builtin_amdgcn_wmma_f32_16x16x32_f16(false, a1, false, b0, (short)0, c10, false, false);
        c11 = __builtin_amdgcn_wmma_f32_16x16x32_f16(false, a1, false, b1, (short)0, c11, false, false);
    }
    // epilogue: 4 tiles of 16x16
#pragma unroll
    for (int mi = 0; mi < 2; ++mi) {
#pragma unroll
        for (int ni = 0; ni < 2; ++ni) {
            v8f c = (mi == 0) ? ((ni == 0) ? c00 : c01) : ((ni == 0) ? c10 : c11);
#pragma unroll
            for (int i = 0; i < 8; ++i) {
                int row = m0 + wm * 32 + mi * 16 + grp * 8 + i;
                int col = n0 + wn * 32 + ni * 16 + lr;
                float v = c[i] + bias[col];
                if (outF) outF[(size_t)row * N + col] = v;
                if (outH) outH[(size_t)row * N + col] = (_Float16)v;
            }
        }
    }
}

// ---------------------------------------------------------------------------
// Attention scores: scores[b,h,q,k] = (Q.K^T)/8 + prefixLM/pad mask bias
// One wave per 16x16 tile; K-dim = DH = 64 (2 WMMA steps)
// ---------------------------------------------------------------------------
__global__ __launch_bounds__(32) void k_attn_scores(const _Float16* __restrict__ q16,
                                                    const _Float16* __restrict__ k16,
                                                    const int* __restrict__ x,
                                                    const int* __restrict__ sep,
                                                    float* __restrict__ scores)
{
    int q0 = blockIdx.x * 16, k0 = blockIdx.y * 16;
    int z = blockIdx.z; int b = z / NHn, head = z % NHn;
    int lane = threadIdx.x; int grp = lane >> 4, lr = lane & 15;
    const _Float16* Q  = q16 + (size_t)b * Sn * Hn + head * DHn;
    const _Float16* Kp = k16 + (size_t)b * Sn * Hn + head * DHn;
    v8f c = {};
#pragma unroll
    for (int d0 = 0; d0 < DHn; d0 += 32) {
        v16h a, bb;
#pragma unroll
        for (int j = 0; j < 16; ++j) {
            int vg = j >> 1;
            int da = d0 + ((vg < 4) ? 0 : 16) + ((vg & 3) * 2) + (j & 1) + grp * 8;
            a[j]  = Q[(size_t)(q0 + lr) * Hn + da];
            int db = d0 + ((j < 8) ? j : j + 8) + grp * 8;
            bb[j] = Kp[(size_t)(k0 + lr) * Hn + db];
        }
        c = __builtin_amdgcn_wmma_f32_16x16x32_f16(false, a, false, bb, (short)0, c, false, false);
    }
    int sp = sep[b];
#pragma unroll
    for (int i = 0; i < 8; ++i) {
        int qp = q0 + grp * 8 + i;
        int kp = k0 + lr;
        float m = ((qp < sp) && (kp < sp)) ? 1.f : ((qp >= kp) ? 1.f : 0.f);
        if (x[b * Sn + kp] == PADtok) m = 0.f;
        scores[((size_t)z * Sn + qp) * Sn + kp] = c[i] * 0.125f + (1.f - m) * -1e9f;
    }
}

// ---------------------------------------------------------------------------
// Row softmax over S=256, f16 probs output
// ---------------------------------------------------------------------------
__global__ __launch_bounds__(256) void k_softmax(const float* __restrict__ scores,
                                                 _Float16* __restrict__ probs)
{
    __shared__ float red[256];
    int qrow = blockIdx.x; int z = blockIdx.y;
    const float* row = scores + ((size_t)z * Sn + qrow) * Sn;
    int t = threadIdx.x;
    float v = row[t];
    red[t] = v; __syncthreads();
    for (int s = 128; s > 0; s >>= 1) { if (t < s) red[t] = fmaxf(red[t], red[t + s]); __syncthreads(); }
    float mx = red[0]; __syncthreads();
    float e = __expf(v - mx);
    red[t] = e; __syncthreads();
    for (int s = 128; s > 0; s >>= 1) { if (t < s) red[t] += red[t + s]; __syncthreads(); }
    float inv = 1.f / red[0];
    probs[((size_t)z * Sn + qrow) * Sn + t] = (_Float16)(e * inv);
}

// ---------------------------------------------------------------------------
// ctx[b,q,h*64+d] = probs @ V ; one wave per 16x16 tile, K-dim = S (8 WMMA steps)
// ---------------------------------------------------------------------------
__global__ __launch_bounds__(32) void k_attn_ctx(const _Float16* __restrict__ probs,
                                                 const _Float16* __restrict__ v16f,
                                                 _Float16* __restrict__ ctx16)
{
    int d0 = blockIdx.x * 16, q0 = blockIdx.y * 16;
    int z = blockIdx.z; int b = z / NHn, head = z % NHn;
    int lane = threadIdx.x; int grp = lane >> 4, lr = lane & 15;
    const _Float16* P  = probs + (size_t)z * Sn * Sn;
    const _Float16* Vp = v16f + (size_t)b * Sn * Hn + head * DHn + d0;
    v8f c = {};
    for (int kk = 0; kk < Sn; kk += 32) {
        v16h a, bb;
#pragma unroll
        for (int j = 0; j < 16; ++j) {
            int vg = j >> 1;
            int ka = kk + ((vg < 4) ? 0 : 16) + ((vg & 3) * 2) + (j & 1) + grp * 8;
            a[j]  = P[(size_t)(q0 + lr) * Sn + ka];
            int kb = kk + ((j < 8) ? j : j + 8) + grp * 8;
            bb[j] = Vp[(size_t)kb * Hn + lr];
        }
        c = __builtin_amdgcn_wmma_f32_16x16x32_f16(false, a, false, bb, (short)0, c, false, false);
    }
#pragma unroll
    for (int i = 0; i < 8; ++i) {
        int qp = q0 + grp * 8 + i;
        ctx16[((size_t)b * Sn + qp) * Hn + head * DHn + d0 + lr] = (_Float16)c[i];
    }
}

// ---------------------------------------------------------------------------
// exact GELU, f32 -> f16
// ---------------------------------------------------------------------------
__global__ __launch_bounds__(256) void k_gelu(const float* __restrict__ in,
                                              _Float16* __restrict__ out16, size_t n)
{
    size_t i = (size_t)blockIdx.x * 256 + threadIdx.x;
    if (i < n) {
        float xv = in[i];
        float g = 0.5f * xv * (1.f + erff(xv * 0.70710678118f));
        out16[i] = (_Float16)g;
    }
}

// ---------------------------------------------------------------------------
// Streaming classifier + CE: 16 token rows per block, 128-col vocab chunks,
// WMMA logits tiles + online logsumexp; never materializes full logits.
// ---------------------------------------------------------------------------
__global__ __launch_bounds__(256) void k_cls_ce(const _Float16* __restrict__ h16,
                                                const float* __restrict__ W,
                                                const float* __restrict__ bias,
                                                const int* __restrict__ y,
                                                float* __restrict__ nll)
{
    __shared__ _Float16 Ah[16][Hn];      // 24 KB
    __shared__ float lg[16][128];        // 8 KB
    __shared__ float mrun[16], srun[16], tgt[16];
    int r0 = blockIdx.x * 16;
    int t = threadIdx.x;
    int wid = t >> 5, lane = t & 31, grp = lane >> 4, lr = lane & 15;
    for (int idx = t; idx < 16 * Hn; idx += 256) {
        int r = idx / Hn, cc = idx % Hn;
        Ah[r][cc] = h16[(size_t)(r0 + r) * Hn + cc];
    }
    if (t < 16) { mrun[t] = -1e30f; srun[t] = 0.f; tgt[t] = 0.f; }
    __syncthreads();
    const int NCH = (Vn + 127) / 128;
    for (int ch = 0; ch < NCH; ++ch) {
        int c0 = ch * 128 + wid * 16;
        int col = c0 + lr;
        v8f c = {};
        for (int k0 = 0; k0 < Hn; k0 += 32) {
            v16h a, bb;
#pragma unroll
            for (int j = 0; j < 16; ++j) {
                int vg = j >> 1;
                int ka = k0 + ((vg < 4) ? 0 : 16) + ((vg & 3) * 2) + (j & 1) + grp * 8;
                a[j] = Ah[lr][ka];
                int kb = k0 + ((j < 8) ? j : j + 8) + grp * 8;
                bb[j] = (col < Vn) ? (_Float16)W[(size_t)kb * Vn + col] : (_Float16)0.f;
            }
            c = __builtin_amdgcn_wmma_f32_16x16x32_f16(false, a, false, bb, (short)0, c, false, false);
        }
#pragma unroll
        for (int i = 0; i < 8; ++i) {
            int rr = grp * 8 + i;
            float v = (col < Vn) ? (c[i] + bias[col]) : -1e30f;
            lg[rr][wid * 16 + lr] = v;
        }
        __syncthreads();
        if (t < 16) {
            int base = ch * 128;
            float m = mrun[t], sacc = srun[t];
            float cm = -1e30f;
            for (int j = 0; j < 128; ++j) cm = fmaxf(cm, lg[t][j]);
            float nm = fmaxf(m, cm);
            sacc *= __expf(m - nm);
            for (int j = 0; j < 128; ++j) sacc += __expf(lg[t][j] - nm);
            mrun[t] = nm; srun[t] = sacc;
            int tcol = y[r0 + t];
            if (tcol >= base && tcol < base + 128) tgt[t] = lg[t][tcol - base];
        }
        __syncthreads();
    }
    if (t < 16) nll[r0 + t] = (mrun[t] + __logf(srun[t])) - tgt[t];
}

__global__ __launch_bounds__(256) void k_reduce(const float* __restrict__ nll, float* __restrict__ out)
{
    __shared__ float red[256];
    int t = threadIdx.x;
    float sum = 0.f;
    for (int i = t; i < Mtok; i += 256) sum += nll[i];
    red[t] = sum; __syncthreads();
    for (int st = 128; st > 0; st >>= 1) { if (t < st) red[t] += red[t + st]; __syncthreads(); }
    if (t == 0) out[0] = red[0] / (float)Mtok;
}

// ---------------------------------------------------------------------------
extern "C" void kernel_launch(void* const* d_in, const int* in_sizes, int n_in,
                              void* d_out, int out_size, void* d_ws, size_t ws_size,
                              hipStream_t stream)
{
    (void)in_sizes; (void)n_in; (void)out_size; (void)ws_size;
    const int*   x        = (const int*)d_in[1];
    const int*   y        = (const int*)d_in[2];
    const float* word_emb = (const float*)d_in[3];
    const float* pos_emb  = (const float*)d_in[4];
    const float* type_emb = (const float*)d_in[5];
    const float* eln_s    = (const float*)d_in[6];
    const float* eln_b    = (const float*)d_in[7];
    const float* Wq = (const float*)d_in[8];   const float* bq = (const float*)d_in[9];
    const float* Wk = (const float*)d_in[10];  const float* bk = (const float*)d_in[11];
    const float* Wv = (const float*)d_in[12];  const float* bv = (const float*)d_in[13];
    const float* Wo = (const float*)d_in[14];  const float* bo = (const float*)d_in[15];
    const float* ln1s = (const float*)d_in[16]; const float* ln1b = (const float*)d_in[17];
    const float* Wi = (const float*)d_in[18];  const float* bi = (const float*)d_in[19];
    const float* Wf = (const float*)d_in[20];  const float* bf = (const float*)d_in[21];
    const float* ln2s = (const float*)d_in[22]; const float* ln2b = (const float*)d_in[23];
    const float* clsW = (const float*)d_in[24]; const float* clsb = (const float*)d_in[25];
    float* out = (float*)d_out;

    char* ws = (char*)d_ws;
    size_t off = 0;
    auto alloc = [&](size_t bytes) -> void* {
        void* p = ws + off;
        off = (off + bytes + 255) & ~(size_t)255;
        return p;
    };
    float*    h      = (float*)   alloc((size_t)Mtok * Hn * 4);
    _Float16* h16    = (_Float16*)alloc((size_t)Mtok * Hn * 2);
    _Float16* q16    = (_Float16*)alloc((size_t)Mtok * Hn * 2);
    _Float16* k16    = (_Float16*)alloc((size_t)Mtok * Hn * 2);
    _Float16* v16f   = (_Float16*)alloc((size_t)Mtok * Hn * 2);
    _Float16* ctx16  = (_Float16*)alloc((size_t)Mtok * Hn * 2);
    float*    tmp    = (float*)   alloc((size_t)Mtok * FFn * 4);
    _Float16* ffi16  = (_Float16*)alloc((size_t)Mtok * FFn * 2);
    float*    scores = (float*)   alloc((size_t)Bn * NHn * Sn * Sn * 4);
    _Float16* probs  = (_Float16*)alloc((size_t)Bn * NHn * Sn * Sn * 2);
    int*      sep    = (int*)     alloc(256);
    float*    nllb   = (float*)   alloc((size_t)Mtok * 4);

    k_sep<<<1, 32, 0, stream>>>(x, sep);
    k_embed<<<Mtok, 256, 0, stream>>>(x, word_emb, pos_emb, type_emb, eln_s, eln_b, h, h16);

    dim3 gH(Hn / 64, Mtok / 128);    // N=768 : 12 x 16
    dim3 gFF(FFn / 64, Mtok / 128);  // N=3072: 48 x 16
    dim3 gScore(Sn / 16, Sn / 16, Bn * NHn);
    dim3 gSm(Sn, Bn * NHn);
    dim3 gCtx(DHn / 16, Sn / 16, Bn * NHn);

    for (int i = 0; i < Ln; ++i) {
        const float* wq = Wq + (size_t)i * Hn * Hn;  const float* bqi = bq + (size_t)i * Hn;
        const float* wk = Wk + (size_t)i * Hn * Hn;  const float* bki = bk + (size_t)i * Hn;
        const float* wv = Wv + (size_t)i * Hn * Hn;  const float* bvi = bv + (size_t)i * Hn;
        const float* wo = Wo + (size_t)i * Hn * Hn;  const float* boi = bo + (size_t)i * Hn;
        const float* wi = Wi + (size_t)i * Hn * FFn; const float* bii = bi + (size_t)i * FFn;
        const float* wf = Wf + (size_t)i * FFn * Hn; const float* bfi = bf + (size_t)i * Hn;

        k_gemm<<<gH, 256, 0, stream>>>(h16, wq, bqi, nullptr, q16, Mtok, Hn, Hn);
        k_gemm<<<gH, 256, 0, stream>>>(h16, wk, bki, nullptr, k16, Mtok, Hn, Hn);
        k_gemm<<<gH, 256, 0, stream>>>(h16, wv, bvi, nullptr, v16f, Mtok, Hn, Hn);
        k_attn_scores<<<gScore, 32, 0, stream>>>(q16, k16, x, sep, scores);
        k_softmax<<<gSm, 256, 0, stream>>>(scores, probs);
        k_attn_ctx<<<gCtx, 32, 0, stream>>>(probs, v16f, ctx16);
        k_gemm<<<gH, 256, 0, stream>>>(ctx16, wo, boi, tmp, nullptr, Mtok, Hn, Hn);
        k_add_ln<<<Mtok, 256, 0, stream>>>(h, tmp, ln1s + (size_t)i * Hn, ln1b + (size_t)i * Hn, h, h16);
        k_gemm<<<gFF, 256, 0, stream>>>(h16, wi, bii, tmp, nullptr, Mtok, FFn, Hn);
        k_gelu<<<(Mtok * FFn) / 256, 256, 0, stream>>>(tmp, ffi16, (size_t)Mtok * FFn);
        k_gemm<<<gH, 256, 0, stream>>>(ffi16, wf, bfi, tmp, nullptr, Mtok, Hn, FFn);
        k_add_ln<<<Mtok, 256, 0, stream>>>(h, tmp, ln2s + (size_t)i * Hn, ln2b + (size_t)i * Hn, h, h16);
    }

    k_cls_ce<<<Mtok / 16, 256, 0, stream>>>(h16, clsW, clsb, y, nllb);
    k_reduce<<<1, 256, 0, stream>>>(nllb, out);
}
